// AWARE_31825707663898
// MI455X (gfx1250) — matrix operand order repaired and checked by reading, not compiled, not run on hardware
//
#include <hip/hip_runtime.h>

typedef __attribute__((ext_vector_type(16))) __bf16 v16bf;
typedef __attribute__((ext_vector_type(8)))  float  v8f;

#define AW_B   64
#define AW_N   512
#define AW_EMB 256
#define AW_RP  256
#define AW_L   6
#define AW_HID 1536
#define AW_OUT 128
#define AW_BN  (AW_B * AW_N)

#define TM 64
#define TK 32

__device__ __forceinline__ unsigned short f2bf(float f) {
    union { float f; unsigned int u; } v; v.f = f;
    unsigned int u = v.u;
    unsigned int r = u + 0x7FFFu + ((u >> 16) & 1u);   // round-to-nearest-even
    return (unsigned short)(r >> 16);
}
__device__ __forceinline__ float bf2f(unsigned short h) {
    union { unsigned int u; float f; } v; v.u = ((unsigned int)h) << 16;
    return v.f;
}
__device__ __forceinline__ float silu_f(float x) {
    return x / (1.0f + __expf(-x));
}

__device__ __forceinline__ void async_ld16(void* lds, const void* g) {
    // GLOBAL_LOAD_ASYNC_TO_LDS_B128 (ASYNCcnt): LDS-direct DMA, no VGPR trip.
    asm volatile("global_load_async_to_lds_b128 %0, %1, off"
                 :: "v"((unsigned)(unsigned long long)lds),
                    "v"((unsigned long long)g)
                 : "memory");
}

// ---------------------------------------------------------------------------
// Batched NT GEMM: C[b] = epilogue(A[b] @ BT[b]^T + bias)
//   A  : [M x K]  bf16 row-major;  BT : [N x K] bf16 row-major (== B^T)
//   act : 0 none, 1 SiLU, 2 adjacency mask (x = mask? x : -1e8)
//   fuse: 3 -> x = (e0[o] + x) * e1[o]        (walk-state update)
//   Outputs (any subset): out32 fp32, outbf bf16 row-major,
//   outbfT bf16 transposed in 512-row chunks (FnT layout).
// NT = 16x16 WMMA tiles per wave in N; block tile = 64(M) x NT*32(N).
// 2-deep LDS ping-pong fed by async global->LDS loads overlapped with WMMA.
// ---------------------------------------------------------------------------
template <int NT>
__global__ __launch_bounds__(256)
void k_gemm_nt(const unsigned short* __restrict__ A,
               const unsigned short* __restrict__ BT,
               const float* __restrict__ bias,
               const int* __restrict__ mask,
               const float* __restrict__ e0,
               const float* __restrict__ e1,
               float* __restrict__ out32,
               unsigned short* __restrict__ outbf,
               unsigned short* __restrict__ outbfT,
               int M, int Ncols, int K,
               long long sA, long long sBT, long long sC,
               int act, int fuse)
{
    constexpr int TNv   = NT * 32;        // block N tile
    constexpr int QB    = NT / 2;         // B-tile loads per thread
    constexpr int NWAIT = 1 + QB;         // async loads per wave per tile

    __shared__ __align__(16) unsigned short lsA[2][TM * TK];
    __shared__ __align__(16) unsigned short lsB[2][TNv * TK];

    const int b = blockIdx.z;
    const unsigned short* Ab = A  + (long long)b * sA;
    const unsigned short* Bb = BT + (long long)b * sBT;

    const int tid   = threadIdx.x;
    const int lane  = tid & 31;
    const int wid   = tid >> 5;
    const int waveM = wid & 3;
    const int waveN = wid >> 2;
    const int tileM = blockIdx.y * TM;
    const int tileN = blockIdx.x * TNv;
    const int hl    = lane >> 4;
    const int l15   = lane & 15;

    const int arow = tid >> 2, ach = tid & 3;   // A: 64 rows x 4 chunks
    const int br0  = tid >> 2, bc0 = tid & 3;   // B: rows br0 + q*64

    auto stage = [&](int kb, int buf) {
        async_ld16(&lsA[buf][arow * TK + ach * 8],
                   Ab + (size_t)(tileM + arow) * K + kb + ach * 8);
#pragma unroll
        for (int q = 0; q < QB; ++q) {
            int r2 = br0 + q * 64;
            async_ld16(&lsB[buf][r2 * TK + bc0 * 8],
                       Bb + (size_t)(tileN + r2) * K + kb + bc0 * 8);
        }
    };

    v8f acc[NT] = {};
    const int nt = K / TK;

    stage(0, 0);

    for (int it = 0; it < nt; ++it) {
        const int cur = it & 1;
        if (it + 1 < nt) {
            stage(it + 1, cur ^ 1);               // keep next tile in flight
            asm volatile("s_wait_asynccnt %0" :: "i"(NWAIT) : "memory");
        } else {
            asm volatile("s_wait_asynccnt 0x0" ::: "memory");
        }
        __syncthreads();   // tile `it` visible to all waves

        union { v16bf v; unsigned int u[8]; } af;
        const int am = waveM * 16 + l15;
#pragma unroll
        for (int t = 0; t < 8; ++t) {
            int kk = hl * 16 + 2 * t;
            af.u[t] = *(const unsigned int*)&lsA[cur][am * TK + kk];
        }
#pragma unroll
        for (int j = 0; j < NT; ++j) {
            union { v16bf v; unsigned int u[8]; } bfr;
            const int bn = waveN * (NT * 16) + j * 16 + l15;
#pragma unroll
            for (int t = 0; t < 8; ++t) {
                int kk = hl * 16 + 2 * t;
                bfr.u[t] = *(const unsigned int*)&lsB[cur][bn * TK + kk];
            }
            acc[j] = __builtin_amdgcn_wmma_f32_16x16x32_bf16(
                false, af.v, false, bfr.v, (short)0, acc[j], false, false);
        }
        __syncthreads();   // buf[cur] reusable at it+1
    }

    // ---- epilogue ----
    const int m0 = tileM + waveM * 16 + hl * 8;   // 8 consecutive rows per lane
#pragma unroll
    for (int j = 0; j < NT; ++j) {
        const int n = tileN + waveN * (NT * 16) + j * 16 + l15;
        const float bv = bias ? bias[n] : 0.0f;
        float xs[8];
#pragma unroll
        for (int v = 0; v < 8; ++v) {
            float x = acc[j][v] + bv;
            size_t o = (size_t)b * sC + (size_t)(m0 + v) * Ncols + n;
            if (act == 1) x = silu_f(x);
            else if (act == 2) { if (mask[o] == 0) x = -1.0e8f; }
            if (fuse == 3) x = (e0[o] + x) * e1[o];
            xs[v] = x;
            if (out32) out32[o] = x;
            if (outbf) outbf[o] = f2bf(x);
        }
        if (outbfT) {
            // transposed in 512-row chunks: [chunk][n][m], lane rows contiguous
            const int bb = m0 >> 9, mm = m0 & 511;
            size_t oT = (size_t)b * sC + (size_t)bb * Ncols * AW_N
                      + (size_t)n * AW_N + mm;
            uint4 pk;
            unsigned int* pu = (unsigned int*)&pk;
#pragma unroll
            for (int q = 0; q < 4; ++q)
                pu[q] = (unsigned int)f2bf(xs[2 * q])
                      | ((unsigned int)f2bf(xs[2 * q + 1]) << 16);
            *(uint4*)(outbfT + oT) = pk;
        }
    }
}

// fp32 -> bf16 elementwise
__global__ void k_cvt(const float* __restrict__ src, unsigned short* __restrict__ dst, int n) {
    int i = blockIdx.x * 256 + threadIdx.x;
    if (i < n) dst[i] = f2bf(src[i]);
}

// LDS-tiled transpose+convert: dst[b][c][r] = src[b][r][c]; grid (C/32,R/32,b), block (32,8)
__global__ void k_tr32(const float* __restrict__ src, unsigned short* __restrict__ dst,
                       int R, int C) {
    __shared__ float t[32][33];
    const int b = blockIdx.z;
    const int c0 = blockIdx.x * 32, r0 = blockIdx.y * 32;
    const size_t base = (size_t)b * R * C;
#pragma unroll
    for (int q = 0; q < 4; ++q) {
        int r = r0 + threadIdx.y + q * 8;
        t[threadIdx.y + q * 8][threadIdx.x] = src[base + (size_t)r * C + c0 + threadIdx.x];
    }
    __syncthreads();
#pragma unroll
    for (int q = 0; q < 4; ++q) {
        int c = c0 + threadIdx.y + q * 8;
        dst[base + (size_t)c * R + r0 + threadIdx.x] = f2bf(t[threadIdx.x][threadIdx.y + q * 8]);
    }
}

// softmax over axis i of pre-masked S[b][i][j]; online max/sum, then write bf16
__global__ void k_softmax(const float* __restrict__ S, unsigned short* __restrict__ AS) {
    int b = blockIdx.x;
    int j = blockIdx.y * 256 + threadIdx.x;
    const float* s = S + (size_t)b * AW_N * AW_N + j;
    unsigned short* o = AS + (size_t)b * AW_N * AW_N + j;
    float mx = -3.0e38f, sum = 0.0f;
    for (int i = 0; i < AW_N; ++i) {
        float v = s[(size_t)i * AW_N];
        float nm = fmaxf(mx, v);
        sum = sum * __expf(mx - nm) + __expf(v - nm);
        mx = nm;
    }
    float inv = 1.0f / sum;
    for (int i = 0; i < AW_N; ++i)
        o[(size_t)i * AW_N] = f2bf(__expf(s[(size_t)i * AW_N] - mx) * inv);
}

// f[b][term*RP + r] = sum_i H[b][i][r]
__global__ void k_colsum(const unsigned short* __restrict__ H, float* __restrict__ f, int term) {
    int b = blockIdx.x, r = threadIdx.x;
    const unsigned short* h = H + (size_t)b * AW_N * AW_RP + r;
    float s = 0.0f;
    for (int i = 0; i < AW_N; ++i) s += bf2f(h[(size_t)i * AW_RP]);
    f[b * AW_HID + term * AW_RP + r] = s;
}

// row L2-normalize f[64][1536] -> bf16
__global__ void k_normalize(const float* __restrict__ f, unsigned short* __restrict__ fbf) {
    __shared__ float red[256];
    int b = blockIdx.x, tid = threadIdx.x;
    float ss = 0.0f;
    for (int i = tid; i < AW_HID; i += 256) { float v = f[b * AW_HID + i]; ss += v * v; }
    red[tid] = ss; __syncthreads();
    for (int s = 128; s > 0; s >>= 1) {
        if (tid < s) red[tid] += red[tid + s];
        __syncthreads();
    }
    float scale = 1.0f / fmaxf(sqrtf(red[0]), 1e-12f);
    for (int i = tid; i < AW_HID; i += 256)
        fbf[b * AW_HID + i] = f2bf(f[b * AW_HID + i] * scale);
}

extern "C" void kernel_launch(void* const* d_in, const int* in_sizes, int n_in,
                              void* d_out, int out_size, void* d_ws, size_t ws_size,
                              hipStream_t stream) {
    const float* node = (const float*)d_in[0];
    const int*   adj  = (const int*)d_in[1];
    const float* Wv_w = (const float*)d_in[3];  const float* Wv_b = (const float*)d_in[4];
    const float* Ww_w = (const float*)d_in[5];  const float* Ww_b = (const float*)d_in[6];
    const float* Wg_w = (const float*)d_in[7];  const float* Wg_b = (const float*)d_in[8];
    const float* W0 = (const float*)d_in[9];    const float* b0 = (const float*)d_in[10];
    const float* W1 = (const float*)d_in[11];   const float* b1 = (const float*)d_in[12];
    const float* W2 = (const float*)d_in[13];   const float* b2 = (const float*)d_in[14];
    const float* W3 = (const float*)d_in[15];   const float* b3 = (const float*)d_in[16];
    float* out = (float*)d_out;

    char* base = (char*)d_ws;
    size_t off = 0;
    auto alloc = [&](size_t bytes) -> void* {
        off = (off + 255) & ~(size_t)255;
        void* p = base + off;
        off += bytes;
        return p;
    };
    unsigned short* WvT  = (unsigned short*)alloc((size_t)AW_EMB * AW_RP * 2);
    unsigned short* WwT  = (unsigned short*)alloc((size_t)AW_RP * AW_RP * 2);
    unsigned short* WgT  = (unsigned short*)alloc((size_t)AW_RP * AW_RP * 2);
    unsigned short* W0T  = (unsigned short*)alloc((size_t)AW_HID * AW_HID * 2);
    unsigned short* W1T  = (unsigned short*)alloc((size_t)AW_HID * AW_HID * 2);
    unsigned short* W2T  = (unsigned short*)alloc((size_t)(AW_HID / 2) * AW_HID * 2);
    unsigned short* W3T  = (unsigned short*)alloc((size_t)AW_OUT * (AW_HID / 2) * 2);
    unsigned short* nodeB= (unsigned short*)alloc((size_t)AW_BN * AW_EMB * 2);
    float*          F1   = (float*)alloc((size_t)AW_BN * AW_RP * 4);
    float*          Fn   = (float*)alloc((size_t)AW_BN * AW_RP * 4);
    unsigned short* FnB  = (unsigned short*)alloc((size_t)AW_BN * AW_RP * 2);
    unsigned short* FnTa = (unsigned short*)alloc((size_t)AW_BN * AW_RP * 2);
    unsigned short* FnTb = (unsigned short*)alloc((size_t)AW_BN * AW_RP * 2);
    unsigned short* GB   = (unsigned short*)alloc((size_t)AW_BN * AW_RP * 2);
    float*          S    = (float*)alloc((size_t)AW_B * AW_N * AW_N * 4);
    unsigned short* ASB  = (unsigned short*)alloc((size_t)AW_B * AW_N * AW_N * 2);
    unsigned short* HB   = (unsigned short*)alloc((size_t)AW_BN * AW_RP * 2);
    float*          fbuf = (float*)alloc((size_t)AW_B * AW_HID * 4);
    unsigned short* fB   = (unsigned short*)alloc((size_t)AW_B * AW_HID * 2);
    unsigned short* h0   = (unsigned short*)alloc((size_t)AW_B * AW_HID * 2);
    unsigned short* h1   = (unsigned short*)alloc((size_t)AW_B * AW_HID * 2);
    unsigned short* h2   = (unsigned short*)alloc((size_t)AW_B * (AW_HID / 2) * 2);

    auto gemm = [&](const unsigned short* A, const unsigned short* BT, const float* bias,
                    const int* mask, const float* e0, const float* e1,
                    float* out32, unsigned short* outbf, unsigned short* outbfT,
                    int M, int Nc, int K,
                    long long sA, long long sBT, long long sC,
                    int batch, int act, int fuse) {
        if (Nc % 256 == 0) {
            dim3 g(Nc / 256, M / TM, batch);
            k_gemm_nt<8><<<g, 256, 0, stream>>>(A, BT, bias, mask, e0, e1,
                out32, outbf, outbfT, M, Nc, K, sA, sBT, sC, act, fuse);
        } else {
            dim3 g(Nc / 128, M / TM, batch);
            k_gemm_nt<4><<<g, 256, 0, stream>>>(A, BT, bias, mask, e0, e1,
                out32, outbf, outbfT, M, Nc, K, sA, sBT, sC, act, fuse);
        }
    };
    auto tr = [&](const float* src, unsigned short* dst, int R, int C, int batch) {
        k_tr32<<<dim3(C / 32, R / 32, batch), dim3(32, 8), 0, stream>>>(src, dst, R, C);
    };

    // --- weight transposes (fp32 -> bf16 [cols x K]) ---
    tr(Wv_w, WvT, AW_EMB, AW_RP, 1);
    tr(Ww_w, WwT, AW_RP, AW_RP, 1);
    tr(Wg_w, WgT, AW_RP, AW_RP, 1);
    tr(W0, W0T, AW_HID, AW_HID, 1);
    tr(W1, W1T, AW_HID, AW_HID, 1);
    tr(W2, W2T, AW_HID, AW_HID / 2, 1);
    tr(W3, W3T, AW_HID / 2, AW_OUT, 1);

    // --- F1 = silu(node @ Wv + b); triple output seeds Fn/FnB/FnT ---
    const int nFeat = AW_BN * AW_EMB;
    k_cvt<<<(nFeat + 255) / 256, 256, 0, stream>>>(node, nodeB, nFeat);
    gemm(nodeB, WvT, Wv_b, nullptr, nullptr, nullptr,
         F1, FnB, FnTa, AW_BN, AW_RP, AW_EMB, 0, 0, 0, 1, /*silu*/1, 0);

    const float* FnCur = F1;
    unsigned short* FnTcur = FnTa;
    unsigned short* FnTnext = FnTb;

    // term 0
    gemm(FnB, WgT, Wg_b, nullptr, nullptr, nullptr,
         nullptr, HB, nullptr, AW_BN, AW_RP, AW_RP, 0, 0, 0, 1, 1, 0);
    k_colsum<<<AW_B, 256, 0, stream>>>(HB, fbuf, 0);

    for (int t = 1; t < AW_L; ++t) {
        // G = Fn @ Ww + b
        gemm(FnB, WwT, Ww_b, nullptr, nullptr, nullptr,
             nullptr, GB, nullptr, AW_BN, AW_RP, AW_RP, 0, 0, 0, 1, 0, 0);
        // S[b] = mask(G[b] @ Fn[b]^T)   (adjacency fused)
        gemm(GB, FnB, nullptr, adj, nullptr, nullptr,
             S, nullptr, nullptr, AW_N, AW_N, AW_RP,
             (long long)AW_N * AW_RP, (long long)AW_N * AW_RP, (long long)AW_N * AW_N,
             AW_B, 2, 0);
        // column softmax -> bf16 A_S
        k_softmax<<<dim3(AW_B, AW_N / 256), 256, 0, stream>>>(S, ASB);
        // Fn' = (Fn + A_S @ Fn) * F1, fused; emits fp32 + bf16 + transposed bf16
        gemm(ASB, FnTcur, nullptr, nullptr, FnCur, F1,
             Fn, FnB, FnTnext, AW_N, AW_RP, AW_N,
             (long long)AW_N * AW_N, (long long)AW_RP * AW_N, (long long)AW_N * AW_RP,
             AW_B, 0, /*fuse*/3);
        { unsigned short* tmp = FnTcur; FnTcur = FnTnext; FnTnext = tmp; }
        FnCur = Fn;
        // walk term t
        gemm(FnB, WgT, Wg_b, nullptr, nullptr, nullptr,
             nullptr, HB, nullptr, AW_BN, AW_RP, AW_RP, 0, 0, 0, 1, 1, 0);
        k_colsum<<<AW_B, 256, 0, stream>>>(HB, fbuf, t);
    }

    // normalize + MLP
    k_normalize<<<AW_B, 256, 0, stream>>>(fbuf, fB);
    gemm(fB, W0T, b0, nullptr, nullptr, nullptr,
         nullptr, h0, nullptr, AW_B, AW_HID, AW_HID, 0, 0, 0, 1, 1, 0);
    gemm(h0, W1T, b1, nullptr, nullptr, nullptr,
         nullptr, h1, nullptr, AW_B, AW_HID, AW_HID, 0, 0, 0, 1, 1, 0);
    gemm(h1, W2T, b2, nullptr, nullptr, nullptr,
         nullptr, h2, nullptr, AW_B, AW_HID / 2, AW_HID, 0, 0, 0, 1, 1, 0);
    gemm(h2, W3T, b3, nullptr, nullptr, nullptr,
         out, nullptr, nullptr, AW_B, AW_OUT, AW_HID / 2, 0, 0, 0, 1, 0, 0);
}
